// OlympusHeteroGNN_55860344652531
// MI455X (gfx1250) — matrix operand order repaired and checked by reading, not compiled, not run on hardware
//
#include <hip/hip_runtime.h>

#define H 128
#define HEADS 4
#define DH 32
#define GF_RELU 1
#define GF_ACC  2

typedef __attribute__((ext_vector_type(16))) __bf16 v16bf;
typedef __attribute__((ext_vector_type(8)))  float  v8f;
typedef __attribute__((ext_vector_type(16))) unsigned short v16us;

__device__ __forceinline__ unsigned short f2bf_rne(float f) {
  unsigned int u = __float_as_uint(f);
  u = u + 0x7FFFu + ((u >> 16) & 1u);
  return (unsigned short)(u >> 16);
}

__device__ __forceinline__ unsigned int enc_ord(float f) {
  unsigned int u = __float_as_uint(f);
  return (u & 0x80000000u) ? ~u : (u | 0x80000000u);
}
__device__ __forceinline__ float dec_ord(unsigned int u) {
  unsigned int b = (u & 0x80000000u) ? (u & 0x7FFFFFFFu) : ~u;
  return __uint_as_float(b);
}

// ---------------- weight fp32 -> bf16, packed in WMMA B-fragment order ----------------
// dst layout per matrix: [ntile(8)][kstep(4)][lane(32)][e(16)] halfs, so each lane's
// 16-element B fragment is 32 contiguous bytes (2x global_load_b128, coalesced).
// B fragment mapping (ISA 16-bit 32x16 B layout): n=(lane&15)+ntile*16,
// k = kstep*32 + (lane<16 ? 0 : 8) + (e<8 ? e : e+8).
__global__ void k_packW(const float* __restrict__ src, unsigned short* __restrict__ dst,
                        int nmat) {
  int idx = blockIdx.x * blockDim.x + threadIdx.x;
  if (idx >= nmat * 16384) return;
  int mat  = idx >> 14;
  int r    = idx & 16383;
  int e    = r & 15;
  int lane = (r >> 4) & 31;
  int kk   = (r >> 9) & 3;
  int nt   = r >> 11;
  int n = (lane & 15) + nt * 16;
  int k = kk * 32 + ((lane < 16) ? 0 : 8) + ((e < 8) ? e : e + 8);
  dst[idx] = f2bf_rne(src[mat * 16384 + k * H + n]);
}

// ---------------- encoder layer 1: relu(x @ W1 + b1), small fan-in ----------------
__global__ void k_enc1(const float* __restrict__ x, const float* __restrict__ W1,
                       const float* __restrict__ b1, float* __restrict__ out,
                       int n, int fi) {
  int idx = blockIdx.x * blockDim.x + threadIdx.x;
  if (idx >= n * H) return;
  int row = idx >> 7, j = idx & 127;
  float s = b1[j];
  const float* xr = x + (long long)row * fi;
  for (int i = 0; i < fi; ++i) s += xr[i] * W1[i * H + j];
  out[idx] = s > 0.f ? s : 0.f;
}

// ---------------- WMMA GEMM: C[N x 128] = A[N x 128] @ W[128 x 128] ----------------
// 32 rows per block (two 16-row M tiles per wave), 8 waves cover N=128.
// A fp32 -> bf16 staged in LDS; W pre-packed in fragment order (see k_packW).
// flags: GF_ACC -> C += result ; GF_RELU -> relu at end ; bias may be null.
__global__ __launch_bounds__(256) void k_gemm128(
    const float* __restrict__ A, const unsigned short* __restrict__ Wf,
    const float* __restrict__ bias, float* __restrict__ C, int nrows, int flags) {
  __shared__ unsigned short Atile[32 * 136];   // padded stride vs LDS banks
  int tid = threadIdx.x;
  int rowBase = blockIdx.x * 32;

  // cooperative load/convert of 32x128 fp32 -> bf16 LDS tile
  #pragma unroll
  for (int t = 0; t < 16; ++t) {
    int j = tid + t * 256;          // 0..4095
    int r = j >> 7, c = j & 127;
    int gr = rowBase + r;
    float v = (gr < nrows) ? A[(long long)gr * H + c] : 0.f;
    Atile[r * 136 + c] = f2bf_rne(v);
  }
  __syncthreads();

  int wave = tid >> 5;              // 0..7 -> which 16-col N tile
  int lane = tid & 31;
  int m  = lane & 15;
  int lo = (lane < 16) ? 1 : 0;
  int col = (lane & 15) + wave * 16;
  const v16us* Wfrag = (const v16us*)(Wf) + (wave * 4) * 32 + lane;

  v8f acc0 = {};
  v8f acc1 = {};
  #pragma unroll
  for (int kk = 0; kk < 4; ++kk) {
    int K0 = kk * 32;
    int ka = K0 + (lo ? 0 : 8);     // A fragment base K (ISA 16-bit A layout)
    v16us au0, au1;
    #pragma unroll
    for (int e = 0; e < 16; ++e) {
      int kA = (e < 8) ? (ka + e) : (ka + 8 + e);
      au0[e] = Atile[m * 136 + kA];
      au1[e] = Atile[(m + 16) * 136 + kA];
    }
    v16us bu = Wfrag[kk * 32];      // 32B contiguous per lane, coalesced
    v16bf b = __builtin_bit_cast(v16bf, bu);
    acc0 = __builtin_amdgcn_wmma_f32_16x16x32_bf16(
        false, __builtin_bit_cast(v16bf, au0), false, b, (short)0, acc0, false, false);
    acc1 = __builtin_amdgcn_wmma_f32_16x16x32_bf16(
        false, __builtin_bit_cast(v16bf, au1), false, b, (short)0, acc1, false, false);
  }

  float bcol = bias ? bias[col] : 0.f;
  #pragma unroll
  for (int r = 0; r < 8; ++r) {
    int row = lo ? r : (r + 8);     // C layout: lanes 0-15 -> M=r, 16-31 -> M=r+8
    #pragma unroll
    for (int half = 0; half < 2; ++half) {
      int gr = rowBase + half * 16 + row;
      if (gr < nrows) {
        float v = (half ? acc1[r] : acc0[r]) + bcol;
        long long idx = (long long)gr * H + col;
        if (flags & GF_ACC) v += C[idx];
        if (flags & GF_RELU) v = v > 0.f ? v : 0.f;
        C[idx] = v;
      }
    }
  }
}

// ---------------- segment mean: scatter-add + counts, then divide ----------------
__global__ void k_seg_scatter(const float* __restrict__ x, const int* __restrict__ src,
                              const int* __restrict__ dst, int E,
                              float* __restrict__ agg, float* __restrict__ cnt) {
  long long idx = (long long)blockIdx.x * blockDim.x + threadIdx.x;
  if (idx >= (long long)E * H) return;
  int e = (int)(idx >> 7), j = (int)(idx & 127);
  int s = src[e], d = dst[e];
  atomicAdd(&agg[(long long)d * H + j], x[(long long)s * H + j]);
  if (j == 0) atomicAdd(&cnt[d], 1.f);
}

__global__ void k_seg_div(float* __restrict__ agg, const float* __restrict__ cnt, int n) {
  int idx = blockIdx.x * blockDim.x + threadIdx.x;
  if (idx >= n * H) return;
  float c = cnt[idx >> 7];
  agg[idx] /= (c > 1.f ? c : 1.f);
}

__global__ void k_scale_relu(float* __restrict__ y, int n, float s) {
  int i = blockIdx.x * blockDim.x + threadIdx.x;
  if (i < n) { float v = y[i] * s; y[i] = v > 0.f ? v : 0.f; }
}

// ---------------- GATv2 edge kernels ----------------
__global__ void k_gat_logit(const float* __restrict__ xl, const float* __restrict__ xr,
                            const int* __restrict__ src, const int* __restrict__ dst,
                            const float* __restrict__ att, float* __restrict__ logits,
                            unsigned int* __restrict__ mEnc, int E) {
  int idx = blockIdx.x * blockDim.x + threadIdx.x;
  if (idx >= E * HEADS) return;
  int e = idx >> 2, h = idx & 3;
  int s = src[e], d = dst[e];
  const float* pl = xl + (long long)s * H + h * DH;
  const float* pr = xr + (long long)d * H + h * DH;
  const float* pa = att + h * DH;
  float acc = 0.f;
  #pragma unroll 8
  for (int t = 0; t < DH; ++t) {
    float v = pl[t] + pr[t];
    v = v > 0.f ? v : 0.2f * v;       // leaky_relu(0.2)
    acc += v * pa[t];
  }
  logits[idx] = acc;
  atomicMax(&mEnc[d * HEADS + h], enc_ord(acc));
}

__global__ void k_gat_expsum(float* __restrict__ logits, const unsigned int* __restrict__ mEnc,
                             const int* __restrict__ dst, float* __restrict__ den, int E) {
  int idx = blockIdx.x * blockDim.x + threadIdx.x;
  if (idx >= E * HEADS) return;
  int e = idx >> 2, h = idx & 3;
  int d = dst[e];
  unsigned int u = mEnc[d * HEADS + h];
  float m = (u == 0u) ? 0.f : dec_ord(u);   // empty-segment sentinel -> 0 like reference
  float ex = __expf(logits[idx] - m);
  logits[idx] = ex;
  atomicAdd(&den[d * HEADS + h], ex);
}

__global__ void k_gat_scatter(const float* __restrict__ xl, const float* __restrict__ logits,
                              const float* __restrict__ den, const int* __restrict__ src,
                              const int* __restrict__ dst, float* __restrict__ out, int E) {
  long long idx = (long long)blockIdx.x * blockDim.x + threadIdx.x;
  if (idx >= (long long)E * H) return;
  int e = (int)(idx >> 7), j = (int)(idx & 127);
  int h = j >> 5;
  int s = src[e], d = dst[e];
  float alpha = logits[e * HEADS + h] / (den[d * HEADS + h] + 1e-16f);
  atomicAdd(&out[(long long)d * H + j], xl[(long long)s * H + j] * alpha);
}

__global__ void k_gat_final(float* __restrict__ out, const float* __restrict__ b0,
                            const float* __restrict__ b1, const float* __restrict__ b2,
                            int n, float invn) {
  int idx = blockIdx.x * blockDim.x + threadIdx.x;
  if (idx >= n * H) return;
  int j = idx & 127;
  float v = out[idx];
  if (b0) v += b0[j];
  if (b1) v += b1[j];
  if (b2) v += b2[j];
  out[idx] = v * invn;
}

// ---------------- host orchestration ----------------
extern "C" void kernel_launch(void* const* d_in, const int* in_sizes, int n_in,
                              void* d_out, int out_size, void* d_ws, size_t ws_size,
                              hipStream_t stream) {
  (void)n_in; (void)ws_size;
  const float* xin[4] = {(const float*)d_in[0], (const float*)d_in[1],
                         (const float*)d_in[2], (const float*)d_in[3]};
  const int fi[4] = {7, 3, 384, 2};
  int N[4]; for (int t = 0; t < 4; ++t) N[t] = in_sizes[t] / fi[t];

  const int* eAC = (const int*)d_in[4]; int Eac = in_sizes[4] / 2;
  const int* eAE = (const int*)d_in[5]; int Eae = in_sizes[5] / 2;
  const int* eAG = (const int*)d_in[6]; int Eag = in_sizes[6] / 2;
  const int* eEG = (const int*)d_in[7]; int Eeg = in_sizes[7] / 2;

  const float* W1[4] = {(const float*)d_in[8],  (const float*)d_in[12],
                        (const float*)d_in[16], (const float*)d_in[20]};
  const float* b1[4] = {(const float*)d_in[9],  (const float*)d_in[13],
                        (const float*)d_in[17], (const float*)d_in[21]};
  const float* W2[4] = {(const float*)d_in[10], (const float*)d_in[14],
                        (const float*)d_in[18], (const float*)d_in[22]};
  const float* b2[4] = {(const float*)d_in[11], (const float*)d_in[15],
                        (const float*)d_in[19], (const float*)d_in[23]};

  const float* sWl  = (const float*)d_in[24];
  const float* sbl  = (const float*)d_in[25];
  const float* sWr  = (const float*)d_in[26];
  const float* gWl  = (const float*)d_in[27];
  const float* gbl  = (const float*)d_in[28];
  const float* gWr  = (const float*)d_in[29];
  const float* gbr  = (const float*)d_in[30];
  const float* gatt = (const float*)d_in[31];
  const float* gbias= (const float*)d_in[32];

  // ---- workspace carve-up ----
  char* base = (char*)d_ws;
  size_t off = 0;
  auto carve = [&](size_t bytes) -> char* {
    off = (off + 255) & ~(size_t)255;
    char* p = base + off; off += bytes; return p;
  };
  const size_t MW = 16384;  // elements in one 128x128 matrix
  unsigned short* wbf = (unsigned short*)carve(36 * MW * sizeof(unsigned short));
  float* enc[4]; for (int t = 0; t < 4; ++t) enc[t] = (float*)carve((size_t)N[t] * H * 4);
  float* l1[4];  for (int t = 0; t < 4; ++t) l1[t]  = (float*)carve((size_t)N[t] * H * 4);
  float* tmp1 = (float*)carve((size_t)N[0] * H * 4);      // hidden / agg / xl
  float* tmp2 = (float*)carve((size_t)N[0] * H * 4);      // xr
  float* cnt  = (float*)carve((size_t)N[0] * 4);
  int Emax = Eac; if (Eae > Emax) Emax = Eae; if (Eag > Emax) Emax = Eag; if (Eeg > Emax) Emax = Eeg;
  float* logits = (float*)carve((size_t)Emax * HEADS * 4);
  unsigned int* mEnc = (unsigned int*)carve((size_t)N[0] * HEADS * 4);
  float* den = (float*)carve((size_t)N[0] * HEADS * 4);

  auto cdiv = [](long long a, long long b) { return (unsigned)((a + b - 1) / b); };
  auto gemm = [&](const float* Am, const unsigned short* Wb, const float* bias,
                  float* Cm, int nr, int flags) {
    k_gemm128<<<cdiv(nr, 32), 256, 0, stream>>>(Am, Wb, bias, Cm, nr, flags);
  };

  // ---- stage 0: weights -> bf16, fragment-packed ----
  for (int t = 0; t < 4; ++t)
    k_packW<<<cdiv(MW, 256), 256, 0, stream>>>(W2[t], wbf + t * MW, 1);
  k_packW<<<cdiv(8 * MW, 256), 256, 0, stream>>>(sWl, wbf + 4 * MW,  8);
  k_packW<<<cdiv(8 * MW, 256), 256, 0, stream>>>(sWr, wbf + 12 * MW, 8);
  k_packW<<<cdiv(8 * MW, 256), 256, 0, stream>>>(gWl, wbf + 20 * MW, 8);
  k_packW<<<cdiv(8 * MW, 256), 256, 0, stream>>>(gWr, wbf + 28 * MW, 8);

  // ---- stage 1: encoders ----
  for (int t = 0; t < 4; ++t) {
    k_enc1<<<cdiv((long long)N[t] * H, 256), 256, 0, stream>>>(xin[t], W1[t], b1[t], tmp1, N[t], fi[t]);
    gemm(tmp1, wbf + t * MW, b2[t], enc[t], N[t], 0);
  }

  // ---- edge-type table (k order matches reference etypes) ----
  struct ET { int s, d; const int* src; const int* dst; int E; };
  ET et[8] = {
    {0, 1, eAC,       eAC + Eac, Eac},   // athlete -> country
    {1, 0, eAC + Eac, eAC,       Eac},   // country -> athlete
    {0, 2, eAE,       eAE + Eae, Eae},   // athlete -> event
    {2, 0, eAE + Eae, eAE,       Eae},   // event   -> athlete
    {0, 3, eAG,       eAG + Eag, Eag},   // athlete -> games
    {3, 0, eAG + Eag, eAG,       Eag},   // games   -> athlete
    {2, 3, eEG,       eEG + Eeg, Eeg},   // event   -> games
    {3, 2, eEG + Eeg, eEG,       Eeg},   // games   -> event
  };
  const float ndeg[4] = {3.f, 1.f, 2.f, 2.f};  // contributions per dst type

  // ---- stage 2: hetero SAGE layer ----
  for (int t = 0; t < 4; ++t)
    hipMemsetAsync(l1[t], 0, (size_t)N[t] * H * 4, stream);
  for (int k = 0; k < 8; ++k) {
    int s = et[k].s, d = et[k].d, E = et[k].E, Nd = N[d];
    hipMemsetAsync(tmp1, 0, (size_t)Nd * H * 4, stream);
    hipMemsetAsync(cnt,  0, (size_t)Nd * 4, stream);
    k_seg_scatter<<<cdiv((long long)E * H, 256), 256, 0, stream>>>(enc[s], et[k].src, et[k].dst, E, tmp1, cnt);
    k_seg_div<<<cdiv((long long)Nd * H, 256), 256, 0, stream>>>(tmp1, cnt, Nd);
    gemm(tmp1,   wbf + (4 + k) * MW,  sbl + k * H, l1[d], Nd, GF_ACC);
    gemm(enc[d], wbf + (12 + k) * MW, nullptr,     l1[d], Nd, GF_ACC);
  }
  for (int t = 0; t < 4; ++t)
    k_scale_relu<<<cdiv((long long)N[t] * H, 256), 256, 0, stream>>>(l1[t], N[t] * H, 1.f / ndeg[t]);

  // ---- stage 3: hetero GATv2 layer, accumulated straight into d_out ----
  float* out = (float*)d_out;
  float* outT[4];
  outT[0] = out;
  outT[1] = outT[0] + (size_t)N[0] * H;
  outT[2] = outT[1] + (size_t)N[1] * H;
  outT[3] = outT[2] + (size_t)N[2] * H;
  hipMemsetAsync(d_out, 0, (size_t)out_size * 4, stream);

  for (int k = 0; k < 8; ++k) {
    int s = et[k].s, d = et[k].d, E = et[k].E, Ns = N[s], Nd = N[d];
    gemm(l1[s], wbf + (20 + k) * MW, gbl + k * H, tmp1, Ns, 0);   // xl
    gemm(l1[d], wbf + (28 + k) * MW, gbr + k * H, tmp2, Nd, 0);   // xr
    hipMemsetAsync(mEnc, 0, (size_t)Nd * HEADS * 4, stream);
    hipMemsetAsync(den,  0, (size_t)Nd * HEADS * 4, stream);
    k_gat_logit<<<cdiv((long long)E * HEADS, 256), 256, 0, stream>>>(
        tmp1, tmp2, et[k].src, et[k].dst, gatt + k * H, logits, mEnc, E);
    k_gat_expsum<<<cdiv((long long)E * HEADS, 256), 256, 0, stream>>>(
        logits, mEnc, et[k].dst, den, E);
    k_gat_scatter<<<cdiv((long long)E * H, 256), 256, 0, stream>>>(
        tmp1, logits, den, et[k].src, et[k].dst, outT[d], E);
  }
  // mean over edge types + per-edge-type biases
  k_gat_final<<<cdiv((long long)N[0] * H, 256), 256, 0, stream>>>(
      outT[0], gbias + 1 * H, gbias + 3 * H, gbias + 5 * H, N[0], 1.f / 3.f);
  k_gat_final<<<cdiv((long long)N[1] * H, 256), 256, 0, stream>>>(
      outT[1], gbias + 0 * H, nullptr, nullptr, N[1], 1.f);
  k_gat_final<<<cdiv((long long)N[2] * H, 256), 256, 0, stream>>>(
      outT[2], gbias + 2 * H, gbias + 7 * H, nullptr, N[2], 0.5f);
  k_gat_final<<<cdiv((long long)N[3] * H, 256), 256, 0, stream>>>(
      outT[3], gbias + 4 * H, gbias + 6 * H, nullptr, N[3], 0.5f);
}
